// LSTMModel_25838523253039
// MI455X (gfx1250) — matrix operand order repaired and checked by reading, not compiled, not run on hardware
//
#include <hip/hip_runtime.h>

typedef __bf16 bf16;
typedef __attribute__((ext_vector_type(16))) __bf16 v16bf;
typedef __attribute__((ext_vector_type(8)))  __bf16 v8bf;
typedef __attribute__((ext_vector_type(8)))  float  v8f;

#define B_TOT 2048
#define T_SEQ 512
#define I_DIM 32
#define H_DIM 64
#define BT    16          // batch rows per workgroup -> 128 workgroups
#define K1    96          // layer1 comb width  (I + H)
#define K2    128         // layer2/3 comb width (H + H)

#define W1_ELEMS (256*K1)             // 24576
#define W2_ELEMS (256*K2)             // 32768
#define W3_ELEMS (256*K2)             // 32768
#define W_TOTAL  (W1_ELEMS+W2_ELEMS+W3_ELEMS)   // 90112 bf16 elems in ws

// ---------------- weight packing: 4 gate matrices -> bf16 [256][K] ----------
__global__ void pack_gates_bf16(const float* Wi, const float* Wf,
                                const float* Wo, const float* Wg,
                                bf16* dst, int K) {
  int idx = blockIdx.x * blockDim.x + threadIdx.x;
  if (idx >= 256 * K) return;
  int n = idx / K, k = idx - n * K;
  const float* W = (n < 64) ? Wi : (n < 128) ? Wf : (n < 192) ? Wo : Wg;
  dst[idx] = (bf16)W[(n & 63) * K + k];
}

// ---------------- fast activations ----------------------------------------
__device__ __forceinline__ float fast_rcp(float x) { return __builtin_amdgcn_rcpf(x); }
__device__ __forceinline__ float sigmoid_f(float x) { return fast_rcp(1.0f + __expf(-x)); }
__device__ __forceinline__ float tanh_f(float x)    { return 1.0f - 2.0f * fast_rcp(__expf(2.0f * x) + 1.0f); }

// ---------------- software-pipelined layer group ---------------------------
// One wave-group (4 waves) owns one LSTM layer. Wave w of the group owns the
// full 16-row batch tile and hidden cols cg*16..cg*16+15 of all 4 gates.
// Weight B-fragments (loop-invariant) live in registers for the entire run.
// Tick s: group GRP processes t = s - GRP. Two barriers per tick (all groups
// execute identical barrier sequences).
template<int KD, int GRP>
__device__ __forceinline__ void layer_pipe(
    const bf16* __restrict__ wg,     // packed layer weights [256][KD] (global)
    const float* __restrict__ x,     // x rows for this block (GRP==0 only)
    const bf16* comb,                // [16][KD] LDS source  = [in | h]
    bf16* d0, int o0,                // h dest 0 (pitch KD)
    bf16* d1, int o1,                // h dest 1 (pitch K2) or nullptr
    bf16* xdst,                      // comb1 (GRP==0 only)
    float4 bias, int tid, int cg, int half, int colq)
{
  constexpr int NKB = KD / 32;

  // ---- loop-invariant B fragments: registers for the whole sequence ----
  v16bf wB[4][NKB];
#pragma unroll
  for (int g = 0; g < 4; ++g) {
    const bf16* wr = wg + (g * 64 + cg * 16 + colq) * KD + half * 16;
#pragma unroll
    for (int kb = 0; kb < NKB; ++kb)
      wB[g][kb] = *(const v16bf*)(wr + kb * 32);
  }

  float c[8];
#pragma unroll
  for (int j = 0; j < 8; ++j) c[j] = 0.f;

  const bf16* arow = comb + colq * KD + half * 16;
  const int   hc   = cg * 16 + colq;
  const int   xr   = tid >> 3;
  const int   xc   = (tid & 7) * 4;

  for (int s = 0; s < T_SEQ + 2; ++s) {
    const bool active = (s >= GRP) && (s < T_SEQ + GRP);

    // prefetch next x tile into registers (waves 0..3 are exactly tid<128)
    float4 xf;
    if (GRP == 0 && tid < 128 && (s + 1) < T_SEQ)
      xf = *(const float4*)(x + (size_t)xr * (T_SEQ * I_DIM)
                              + (size_t)(s + 1) * I_DIM + xc);

    bf16 h[8];
    if (active) {
      v8f a0, a1, a2, a3;
#pragma unroll
      for (int j = 0; j < 8; ++j) { a0[j] = bias.x; a1[j] = bias.y; a2[j] = bias.z; a3[j] = bias.w; }
#pragma unroll
      for (int kb = 0; kb < NKB; ++kb) {
        v16bf a;
        v8bf lo = *(const v8bf*)(arow + kb * 32);
        v8bf hi = *(const v8bf*)(arow + kb * 32 + 8);
#pragma unroll
        for (int j = 0; j < 8; ++j) { a[j] = lo[j]; a[8 + j] = hi[j]; }
        a0 = __builtin_amdgcn_wmma_f32_16x16x32_bf16(false, a, false, wB[0][kb], (short)0, a0, false, false);
        a1 = __builtin_amdgcn_wmma_f32_16x16x32_bf16(false, a, false, wB[1][kb], (short)0, a1, false, false);
        a2 = __builtin_amdgcn_wmma_f32_16x16x32_bf16(false, a, false, wB[2][kb], (short)0, a2, false, false);
        a3 = __builtin_amdgcn_wmma_f32_16x16x32_bf16(false, a, false, wB[3][kb], (short)0, a3, false, false);
      }
#pragma unroll
      for (int j = 0; j < 8; ++j) {
        float iv = sigmoid_f(a0[j]);
        float fv = sigmoid_f(a1[j]);
        float ov = sigmoid_f(a2[j]);
        float gv = tanh_f(a3[j]);
        float cn = fv * c[j] + iv * gv;
        c[j] = cn;
        h[j] = (bf16)(ov * tanh_f(cn));
      }
    }
    __syncthreads();                       // barrier 1: reads done

    if (active) {
#pragma unroll
      for (int j = 0; j < 8; ++j) {
        int row = half * 8 + j;            // D layout: vgpr j -> M = j + 8*half
        d0[row * KD + o0 + hc] = h[j];
        if (d1) d1[row * K2 + o1 + hc] = h[j];
      }
    }
    if (GRP == 0 && tid < 128 && (s + 1) < T_SEQ) {
      bf16* d = xdst + xr * K1 + xc;
      d[0] = (bf16)xf.x; d[1] = (bf16)xf.y; d[2] = (bf16)xf.z; d[3] = (bf16)xf.w;
    }
    __syncthreads();                       // barrier 2: writes visible
  }
}

// ---------------- fused pipelined 3-layer LSTM + FC head --------------------
struct LstmParams {
  const float* x;
  const bf16*  wpack;   // W_TOTAL bf16 in workspace: [w1|w2|w3]
  const float* bi1; const float* bf1; const float* bo1; const float* bg1;
  const float* bi2; const float* bf2; const float* bo2; const float* bg2;
  const float* bi3; const float* bf3; const float* bo3; const float* bg3;
  const float* Wfc; const float* bfc;
  float* out;
};

__global__ __launch_bounds__(384) void lstm3_pipe(LstmParams p) {
  __shared__ __align__(32) bf16 combAll[BT * (K1 + 2 * K2)];   // 11 KB
  bf16* comb1 = combAll;                 // [16][96]   = [x_t | h1]
  bf16* comb2 = comb1 + BT * K1;         // [16][128]  = [h1  | h2]
  bf16* comb3 = comb2 + BT * K2;         // [16][128]  = [h2  | h3]

  const int    tid = threadIdx.x;
  const size_t b0  = (size_t)blockIdx.x * BT;
  const float* xblk = p.x + b0 * (T_SEQ * I_DIM);

  // zero all comb buffers
  {
    uint4 z; z.x = z.y = z.z = z.w = 0u;
    uint4* dc = (uint4*)combAll;
    for (int i = tid; i < (BT * (K1 + 2 * K2)) / 8; i += 384) dc[i] = z;
  }
  __syncthreads();
  // stage x[0]
  if (tid < 128) {
    int r = tid >> 3, c0 = (tid & 7) * 4;
    float4 f = *(const float4*)(xblk + (size_t)r * (T_SEQ * I_DIM) + c0);
    bf16* d = comb1 + r * K1 + c0;
    d[0] = (bf16)f.x; d[1] = (bf16)f.y; d[2] = (bf16)f.z; d[3] = (bf16)f.w;
  }
  __syncthreads();

  const int wv   = tid >> 5;     // 0..11
  const int grp  = wv >> 2;      // layer group 0..2
  const int cg   = wv & 3;       // hidden-col group
  const int lane = tid & 31;
  const int half = lane >> 4;
  const int colq = lane & 15;
  const int hc   = cg * 16 + colq;

  float4 bias;
  if (grp == 0)      bias = make_float4(p.bi1[hc], p.bf1[hc], p.bo1[hc], p.bg1[hc]);
  else if (grp == 1) bias = make_float4(p.bi2[hc], p.bf2[hc], p.bo2[hc], p.bg2[hc]);
  else               bias = make_float4(p.bi3[hc], p.bf3[hc], p.bo3[hc], p.bg3[hc]);

  const bf16* w1g = p.wpack;
  const bf16* w2g = p.wpack + W1_ELEMS;
  const bf16* w3g = p.wpack + W1_ELEMS + W2_ELEMS;

  if (grp == 0)
    layer_pipe<K1, 0>(w1g, xblk, comb1, comb1, 32, comb2, 0, comb1,
                      bias, tid, cg, half, colq);
  else if (grp == 1)
    layer_pipe<K2, 1>(w2g, nullptr, comb2, comb2, 64, comb3, 0, nullptr,
                      bias, tid, cg, half, colq);
  else
    layer_pipe<K2, 2>(w3g, nullptr, comb3, comb3, 64, (bf16*)nullptr, 0, nullptr,
                      bias, tid, cg, half, colq);

  __syncthreads();

  // FC head: out[b] = h3[b,:] . Wfc + bfc   (h3 in comb3[:,64:128])
  if (tid < BT) {
    float s = p.bfc[0];
    const bf16* hrow = comb3 + tid * K2 + 64;
#pragma unroll
    for (int h = 0; h < H_DIM; ++h) s += (float)hrow[h] * p.Wfc[h];
    p.out[b0 + tid] = s;
  }
}

// ---------------- launch ----------------------------------------------------
extern "C" void kernel_launch(void* const* d_in, const int* in_sizes, int n_in,
                              void* d_out, int out_size, void* d_ws, size_t ws_size,
                              hipStream_t stream) {
  (void)in_sizes; (void)n_in; (void)out_size; (void)ws_size;

  const float* x = (const float*)d_in[0];
  const float* Wi1 = (const float*)d_in[1];  const float* bi1 = (const float*)d_in[2];
  const float* Wf1 = (const float*)d_in[3];  const float* bf1 = (const float*)d_in[4];
  const float* Wo1 = (const float*)d_in[5];  const float* bo1 = (const float*)d_in[6];
  const float* Wg1 = (const float*)d_in[7];  const float* bg1 = (const float*)d_in[8];
  const float* Wi2 = (const float*)d_in[9];  const float* bi2 = (const float*)d_in[10];
  const float* Wf2 = (const float*)d_in[11]; const float* bf2 = (const float*)d_in[12];
  const float* Wo2 = (const float*)d_in[13]; const float* bo2 = (const float*)d_in[14];
  const float* Wg2 = (const float*)d_in[15]; const float* bg2 = (const float*)d_in[16];
  const float* Wi3 = (const float*)d_in[17]; const float* bi3 = (const float*)d_in[18];
  const float* Wf3 = (const float*)d_in[19]; const float* bf3 = (const float*)d_in[20];
  const float* Wo3 = (const float*)d_in[21]; const float* bo3 = (const float*)d_in[22];
  const float* Wg3 = (const float*)d_in[23]; const float* bg3 = (const float*)d_in[24];
  const float* Wfc = (const float*)d_in[25]; const float* bfc = (const float*)d_in[26];

  bf16* wpack = (bf16*)d_ws;
  bf16* w1 = wpack;
  bf16* w2 = w1 + W1_ELEMS;
  bf16* w3 = w2 + W2_ELEMS;

  pack_gates_bf16<<<(W1_ELEMS + 255) / 256, 256, 0, stream>>>(Wi1, Wf1, Wo1, Wg1, w1, K1);
  pack_gates_bf16<<<(W2_ELEMS + 255) / 256, 256, 0, stream>>>(Wi2, Wf2, Wo2, Wg2, w2, K2);
  pack_gates_bf16<<<(W3_ELEMS + 255) / 256, 256, 0, stream>>>(Wi3, Wf3, Wo3, Wg3, w3, K2);

  LstmParams p;
  p.x = x; p.wpack = wpack;
  p.bi1 = bi1; p.bf1 = bf1; p.bo1 = bo1; p.bg1 = bg1;
  p.bi2 = bi2; p.bf2 = bf2; p.bo2 = bo2; p.bg2 = bg2;
  p.bi3 = bi3; p.bf3 = bf3; p.bo3 = bo3; p.bg3 = bg3;
  p.Wfc = Wfc; p.bfc = bfc;
  p.out = (float*)d_out;

  lstm3_pipe<<<B_TOT / BT, 384, 0, stream>>>(p);
}
